// CenterMemMatchingAEv4_39204461478037
// MI455X (gfx1250) — matrix operand order repaired and checked by reading, not compiled, not run on hardware
//
#include <hip/hip_runtime.h>
#include <hip/hip_bf16.h>

#define TOPK 10

typedef __attribute__((ext_vector_type(16))) _Float16 v16h;
typedef __attribute__((ext_vector_type(8)))  _Float16 v8h;
typedef __attribute__((ext_vector_type(8)))  float    v8f;

// ---------------------------------------------------------------------------
// Fragment loaders: every fragment is two contiguous 16-byte loads.
//
// A (16x32 f16): lane L -> M = L%16, kb = (L<16?0:8);
//   elements e=0..7  -> K = kb + e        (contiguous)
//   elements e=8..15 -> K = 16 + kb + e-8 (contiguous)
// B (32x16 f16): lane L -> N = L%16, same K order as A
// ---------------------------------------------------------------------------
__device__ __forceinline__ v16h frag_from_row_A(const _Float16* __restrict__ row, int lane) {
  int kb = (lane < 16) ? 0 : 8;
  v8h lo = *(const v8h*)(row + kb);
  v8h hi = *(const v8h*)(row + 16 + kb);
  return __builtin_shufflevector(lo, hi, 0, 1, 2, 3, 4, 5, 6, 7, 8, 9, 10, 11, 12, 13, 14, 15);
}

__device__ __forceinline__ v16h frag_from_row_B(const _Float16* __restrict__ row, int lane) {
  int kofs = (lane < 16) ? 0 : 8;
  v8h lo = *(const v8h*)(row + kofs);
  v8h hi = *(const v8h*)(row + 16 + kofs);
  return __builtin_shufflevector(lo, hi, 0, 1, 2, 3, 4, 5, 6, 7, 8, 9, 10, 11, 12, 13, 14, 15);
}

// B fragment from an NHWC f16 activation tensor at (b, yy, xx[lane], ci0..ci0+31).
// yy check is wave-uniform; xx check is the only per-lane guard (one per fragment,
// loop-invariant -> compiler keeps it as an SGPR mask).
__device__ __forceinline__ v16h load_B_hwc(const _Float16* __restrict__ hwc, int b, int H,
                                           int W, int Cpad, int yy, int xx, int ci0,
                                           int lane) {
  v16h z = {};
  if (yy < 0 || yy >= H) return z;
  if (xx < 0 || xx >= W) return z;
  const _Float16* p = hwc + (((size_t)(b * H + yy) * W + xx) * Cpad + ci0);
  return frag_from_row_B(p, lane);
}

// One K-chunk of fragments for the fused-pool conv (A + four B tiles).
__device__ __forceinline__ void load_pool_chunk(const _Float16* __restrict__ hwc,
                                                const _Float16* __restrict__ wrow,
                                                int b, int H, int W, int Cpad,
                                                int y0, int x0, int n, int lane,
                                                int t, int ci0,
                                                v16h& a, v16h& b00, v16h& b01,
                                                v16h& b10, v16h& b11) {
  int dy = t / 3 - 1, dx = t % 3 - 1;
  int yy0 = y0 + dy, yy1 = yy0 + 1;
  int xxa = x0 + n + dx, xxb = xxa + 16;
  int kc = t * Cpad + ci0;
  a   = frag_from_row_A(wrow + kc, lane);
  b00 = load_B_hwc(hwc, b, H, W, Cpad, yy0, xxa, ci0, lane);
  b01 = load_B_hwc(hwc, b, H, W, Cpad, yy0, xxb, ci0, lane);
  b10 = load_B_hwc(hwc, b, H, W, Cpad, yy1, xxa, ci0, lane);
  b11 = load_B_hwc(hwc, b, H, W, Cpad, yy1, xxb, ci0, lane);
}

// One K-chunk for the plain conv (A + one B tile).
__device__ __forceinline__ void load_conv_chunk(const _Float16* __restrict__ hwc,
                                                const _Float16* __restrict__ wrow,
                                                int b, int H, int W, int Cpad,
                                                int y, int x0, int n, int lane,
                                                int t, int ci0, v16h& a, v16h& bm) {
  int dy = t / 3 - 1, dx = t % 3 - 1;
  int kc = t * Cpad + ci0;
  a  = frag_from_row_A(wrow + kc, lane);
  bm = load_B_hwc(hwc, b, H, W, Cpad, y + dy, x0 + n + dx, ci0, lane);
}

// ---------------------------------------------------------------------------
// 0) center-crop 64x64 + zero pad back to 256x256
// ---------------------------------------------------------------------------
__global__ void crop_pad_kernel(const float* __restrict__ x, float* __restrict__ xc,
                                int B, int H, int W, int crop) {
  int idx = blockIdx.x * blockDim.x + threadIdx.x;
  int total = B * H * W;
  if (idx >= total) return;
  int px = idx % W, t = idx / W;
  int py = t % H, b = t / H;
  int sh = (H - crop) / 2, sw = (W - crop) / 2;
  float v = 0.0f;
  if (py >= sh && py < sh + crop && px >= sw && px < sw + crop)
    v = x[(b * H + py) * W + px];
  xc[(b * H + py) * W + px] = v;
}

// ---------------------------------------------------------------------------
// 1) Cin==1 conv3x3 + ReLU + 2x2 maxpool (scalar; K=9, memory bound)
// ---------------------------------------------------------------------------
__global__ void conv3x3_c1_pool_kernel(const float* __restrict__ in,
                                       const float* __restrict__ w,
                                       const float* __restrict__ bias,
                                       float* __restrict__ out,
                                       int B, int Cout, int H, int W) {
  int Wo = W >> 1, Ho = H >> 1;
  int idx = blockIdx.x * blockDim.x + threadIdx.x;
  int total = B * Cout * Ho * Wo;
  if (idx >= total) return;
  int px = idx % Wo; int t = idx / Wo;
  int py = t % Ho; t /= Ho;
  int co = t % Cout; int b = t / Cout;
  const float* wk = w + co * 9;
  float bv = bias[co];
  float best = 0.0f;  // pooled candidates are post-ReLU (>= 0)
#pragma unroll
  for (int dy = 0; dy < 2; ++dy)
#pragma unroll
    for (int dx = 0; dx < 2; ++dx) {
      int y = py * 2 + dy, x = px * 2 + dx;
      float s = bv;
#pragma unroll
      for (int tt = 0; tt < 9; ++tt) {
        int yy = y + tt / 3 - 1, xx = x + tt % 3 - 1;
        if (yy >= 0 && yy < H && xx >= 0 && xx < W)
          s += wk[tt] * in[(b * H + yy) * W + xx];
      }
      best = fmaxf(best, s);
    }
  out[((b * Cout + co) * Ho + py) * Wo + px] = best;
}

// ---------------------------------------------------------------------------
// 2) Layout transforms feeding the WMMA convs
// ---------------------------------------------------------------------------
// weights (Cout,Cin,3,3) f32 -> f16 [Cout][9*Cpad], tap-major K (k = t*Cpad + ci)
__global__ void pack_weights_kernel(const float* __restrict__ w, _Float16* __restrict__ wpk,
                                    int Cout, int Cin, int Cpad) {
  int idx = blockIdx.x * blockDim.x + threadIdx.x;
  int total = Cout * 9 * Cpad;
  if (idx >= total) return;
  int ci = idx % Cpad; int t = (idx / Cpad) % 9; int co = idx / (Cpad * 9);
  float v = (ci < Cin) ? w[(co * Cin + ci) * 9 + t] : 0.0f;
  wpk[(size_t)co * 9 * Cpad + t * Cpad + ci] = (_Float16)v;
}

// (B,Cin,H,W) f32 -> f16 NHWC [B][H][W][Cpad], channel zero-padded
__global__ void nchw_to_hwc_f16_kernel(const float* __restrict__ in, _Float16* __restrict__ out,
                                       int B, int Cin, int Cpad, int H, int W) {
  int idx = blockIdx.x * blockDim.x + threadIdx.x;
  int total = B * H * W * Cpad;
  if (idx >= total) return;
  int c = idx % Cpad; int t = idx / Cpad;
  int x = t % W; t /= W;
  int y = t % H; int b = t / H;
  float v = (c < Cin) ? in[((b * Cin + c) * H + y) * W + x] : 0.0f;
  out[idx] = (_Float16)v;
}

// ---------------------------------------------------------------------------
// 3) WMMA implicit-GEMM conv3x3 + ReLU + fused 2x2 maxpool, software-pipelined:
//    chunk i+1's fragment loads are issued before chunk i's WMMAs so the
//    s_wait before the math covers only old loads.
// ---------------------------------------------------------------------------
__global__ __launch_bounds__(32)
void conv3x3_wmma_pool_kernel(const _Float16* __restrict__ hwc, const _Float16* __restrict__ wpk,
                              const float* __restrict__ bias, float* __restrict__ out,
                              int B, int Cpad, int Cout, int H, int W) {
  int lane = threadIdx.x;
  int Wo = W >> 1, Ho = H >> 1;
  int tx = blockIdx.x;                 // 16 pooled cols = 32 pre-pool cols
  int py = blockIdx.y;                 // pooled row
  int z = blockIdx.z;
  int tilesCo = Cout >> 4;
  int b = z / tilesCo;
  int coBase = (z % tilesCo) << 4;
  int y0 = py * 2;
  int x0 = tx * 32;
  int K = 9 * Cpad;
  int n = lane & 15;
  const _Float16* wrow = wpk + (size_t)(coBase + n) * K;  // A row (M = lane%16)

  v8f c00 = {}, c01 = {}, c10 = {}, c11 = {};
  v16h a0, p00, p01, p10, p11;
  load_pool_chunk(hwc, wrow, b, H, W, Cpad, y0, x0, n, lane, 0, 0, a0, p00, p01, p10, p11);
  int tc = 0, cc = 0;
  for (int kk = 32; kk < K; kk += 32) {
    cc += 32;
    if (cc == Cpad) { cc = 0; ++tc; }
    v16h a1, q00, q01, q10, q11;
    load_pool_chunk(hwc, wrow, b, H, W, Cpad, y0, x0, n, lane, tc, cc, a1, q00, q01, q10, q11);
    c00 = __builtin_amdgcn_wmma_f32_16x16x32_f16(false, a0, false, p00, (short)0, c00, false, false);
    c01 = __builtin_amdgcn_wmma_f32_16x16x32_f16(false, a0, false, p01, (short)0, c01, false, false);
    c10 = __builtin_amdgcn_wmma_f32_16x16x32_f16(false, a0, false, p10, (short)0, c10, false, false);
    c11 = __builtin_amdgcn_wmma_f32_16x16x32_f16(false, a0, false, p11, (short)0, c11, false, false);
    a0 = a1; p00 = q00; p01 = q01; p10 = q10; p11 = q11;
  }
  c00 = __builtin_amdgcn_wmma_f32_16x16x32_f16(false, a0, false, p00, (short)0, c00, false, false);
  c01 = __builtin_amdgcn_wmma_f32_16x16x32_f16(false, a0, false, p01, (short)0, c01, false, false);
  c10 = __builtin_amdgcn_wmma_f32_16x16x32_f16(false, a0, false, p10, (short)0, c10, false, false);
  c11 = __builtin_amdgcn_wmma_f32_16x16x32_f16(false, a0, false, p11, (short)0, c11, false, false);

  int hi = (lane < 16) ? 0 : 8;
#pragma unroll
  for (int r = 0; r < 8; ++r) {
    int m = coBase + hi + r;
    float bv = bias[m];
    // ReLU and max commute: relu(max(rows)+bias) == max over relu'd values
    float v0 = fmaxf(fmaxf(c00[r], c10[r]) + bv, 0.0f);
    float v1 = fmaxf(fmaxf(c01[r], c11[r]) + bv, 0.0f);
    float p0 = fmaxf(v0, __shfl_xor(v0, 1, 32));
    float p1 = fmaxf(v1, __shfl_xor(v1, 1, 32));
    if ((lane & 1) == 0) {
      int px0 = tx * 16 + (n >> 1);
      out[((b * Cout + m) * Ho + py) * Wo + px0]     = p0;
      out[((b * Cout + m) * Ho + py) * Wo + px0 + 8] = p1;
    }
  }
}

// ---------------------------------------------------------------------------
// 4) WMMA implicit-GEMM conv3x3 (+ReLU), no pool (decoder conv), pipelined
// ---------------------------------------------------------------------------
__global__ __launch_bounds__(32)
void conv3x3_wmma_kernel(const _Float16* __restrict__ hwc, const _Float16* __restrict__ wpk,
                         const float* __restrict__ bias, float* __restrict__ out,
                         int B, int Cpad, int Cout, int H, int W, int relu) {
  int lane = threadIdx.x;
  int x0 = blockIdx.x * 16;
  int y = blockIdx.y;
  int z = blockIdx.z;
  int tilesCo = Cout >> 4;
  int b = z / tilesCo;
  int coBase = (z % tilesCo) << 4;
  int K = 9 * Cpad;
  int n = lane & 15;
  const _Float16* wrow = wpk + (size_t)(coBase + n) * K;

  v8f c = {};
  v16h a0, p0;
  load_conv_chunk(hwc, wrow, b, H, W, Cpad, y, x0, n, lane, 0, 0, a0, p0);
  int tc = 0, cc = 0;
  for (int kk = 32; kk < K; kk += 32) {
    cc += 32;
    if (cc == Cpad) { cc = 0; ++tc; }
    v16h a1, q0;
    load_conv_chunk(hwc, wrow, b, H, W, Cpad, y, x0, n, lane, tc, cc, a1, q0);
    c = __builtin_amdgcn_wmma_f32_16x16x32_f16(false, a0, false, p0, (short)0, c, false, false);
    a0 = a1; p0 = q0;
  }
  c = __builtin_amdgcn_wmma_f32_16x16x32_f16(false, a0, false, p0, (short)0, c, false, false);

  int hi = (lane < 16) ? 0 : 8;
#pragma unroll
  for (int r = 0; r < 8; ++r) {
    int m = coBase + hi + r;
    float v = c[r] + bias[m];
    if (relu) v = fmaxf(v, 0.0f);
    out[((b * Cout + m) * H + y) * W + x0 + n] = v;
  }
}

// ---------------------------------------------------------------------------
// 5) L2-normalize: NCHW feature map -> (B, HW, C) f16 rows
// ---------------------------------------------------------------------------
__global__ void l2norm_nchw_kernel(const float* __restrict__ z, _Float16* __restrict__ qn,
                                   int B, int C, int HW) {
  int idx = blockIdx.x * blockDim.x + threadIdx.x;
  if (idx >= B * HW) return;
  int b = idx / HW, n = idx % HW;
  const float* base = z + (size_t)b * C * HW + n;
  float s = 0.0f;
  for (int c = 0; c < C; ++c) { float v = base[c * HW]; s += v * v; }
  float inv = 1.0f / fmaxf(sqrtf(s), 1e-12f);
  _Float16* dst = qn + (size_t)idx * C;
  for (int c = 0; c < C; ++c) dst[c] = (_Float16)(base[c * HW] * inv);
}

// (M, C) f32 rows -> normalized f16 rows
__global__ void l2norm_mem_kernel(const float* __restrict__ mem, _Float16* __restrict__ kn,
                                  int M, int C) {
  int m = blockIdx.x * blockDim.x + threadIdx.x;
  if (m >= M) return;
  const float* src = mem + (size_t)m * C;
  float s = 0.0f;
  for (int c = 0; c < C; ++c) s += src[c] * src[c];
  float inv = 1.0f / fmaxf(sqrtf(s), 1e-12f);
  _Float16* dst = kn + (size_t)m * C;
  for (int c = 0; c < C; ++c) dst[c] = (_Float16)(src[c] * inv);
}

// ---------------------------------------------------------------------------
// 6) Memory matching, top-10 softmax blend (center branch).
//    A fragments (query rows) are invariant over the 50 memory tiles: loaded
//    once into 4 v16h. B tiles are prefetched one tile ahead. Sims go to LDS
//    (801-col pad -> conflict-free per-lane scans), then 16 lanes do
//    top-10 / softmax / weighted mem_values sum straight into f16 NHWC dcat.
// ---------------------------------------------------------------------------
__global__ __launch_bounds__(32)
void match_center_kernel(const _Float16* __restrict__ qn, const _Float16* __restrict__ kn,
                         const float* __restrict__ vals, _Float16* __restrict__ dcat,
                         int B, int NQ, int M, int C) {
  __shared__ float sims[16][801];
  int lane = threadIdx.x;
  int b = blockIdx.y;
  int qBase = blockIdx.x * 16;
  int mrow = lane & 15;
  int hi = (lane < 16) ? 0 : 8;
  const _Float16* qrow = qn + ((size_t)b * NQ + qBase + mrow) * C;

  v16h a0 = frag_from_row_A(qrow + 0, lane);
  v16h a1 = frag_from_row_A(qrow + 32, lane);
  v16h a2 = frag_from_row_A(qrow + 64, lane);
  v16h a3 = frag_from_row_A(qrow + 96, lane);

  const _Float16* krow = kn + (size_t)mrow * C;
  v16h p0 = frag_from_row_B(krow + 0, lane);
  v16h p1 = frag_from_row_B(krow + 32, lane);
  v16h p2 = frag_from_row_B(krow + 64, lane);
  v16h p3 = frag_from_row_B(krow + 96, lane);

  for (int mt = 0; mt < M; mt += 16) {
    v16h q0 = p0, q1 = p1, q2 = p2, q3 = p3;
    if (mt + 16 < M) {
      const _Float16* nrow = kn + (size_t)(mt + 16 + mrow) * C;
      q0 = frag_from_row_B(nrow + 0, lane);
      q1 = frag_from_row_B(nrow + 32, lane);
      q2 = frag_from_row_B(nrow + 64, lane);
      q3 = frag_from_row_B(nrow + 96, lane);
    }
    v8f c = {};
    c = __builtin_amdgcn_wmma_f32_16x16x32_f16(false, a0, false, p0, (short)0, c, false, false);
    c = __builtin_amdgcn_wmma_f32_16x16x32_f16(false, a1, false, p1, (short)0, c, false, false);
    c = __builtin_amdgcn_wmma_f32_16x16x32_f16(false, a2, false, p2, (short)0, c, false, false);
    c = __builtin_amdgcn_wmma_f32_16x16x32_f16(false, a3, false, p3, (short)0, c, false, false);
#pragma unroll
    for (int r = 0; r < 8; ++r) sims[hi + r][mt + mrow] = c[r];
    p0 = q0; p1 = q1; p2 = q2; p3 = q3;
  }
  __syncthreads();

  if (lane < 16) {
    float tv[TOPK];
    int ti[TOPK];
#pragma unroll
    for (int k = 0; k < TOPK; ++k) { tv[k] = -1e30f; ti[k] = 0; }
    for (int m = 0; m < M; ++m) {
      float v = sims[lane][m];
      if (v > tv[TOPK - 1]) {
        int p = TOPK - 1;
        while (p > 0 && tv[p - 1] < v) { tv[p] = tv[p - 1]; ti[p] = ti[p - 1]; --p; }
        tv[p] = v; ti[p] = m;
      }
    }
    float mx = tv[0];
    float wgt[TOPK];
    float sum = 0.0f;
#pragma unroll
    for (int k = 0; k < TOPK; ++k) { wgt[k] = expf(tv[k] - mx); sum += wgt[k]; }
    float invs = 1.0f / sum;
    int n = qBase + lane;
    _Float16* drow = dcat + ((size_t)b * NQ + n) * (2 * C);
    for (int c = 0; c < C; ++c) {
      float acc = 0.0f;
#pragma unroll
      for (int k = 0; k < TOPK; ++k) acc += wgt[k] * vals[(size_t)ti[k] * C + c];
      drow[c] = (_Float16)(acc * invs);
    }
  }
}

// ---------------------------------------------------------------------------
// 7) Memory matching, argmax hard lookup (skip branch) -> channels [C, 2C)
// ---------------------------------------------------------------------------
__global__ __launch_bounds__(32)
void match_argmax_kernel(const _Float16* __restrict__ qn, const _Float16* __restrict__ kn,
                         const float* __restrict__ hard, _Float16* __restrict__ dcat,
                         int B, int NQ, int M, int C) {
  __shared__ float sims[16][801];
  int lane = threadIdx.x;
  int b = blockIdx.y;
  int qBase = blockIdx.x * 16;
  int mrow = lane & 15;
  int hi = (lane < 16) ? 0 : 8;
  const _Float16* qrow = qn + ((size_t)b * NQ + qBase + mrow) * C;

  v16h a0 = frag_from_row_A(qrow + 0, lane);
  v16h a1 = frag_from_row_A(qrow + 32, lane);
  v16h a2 = frag_from_row_A(qrow + 64, lane);
  v16h a3 = frag_from_row_A(qrow + 96, lane);

  const _Float16* krow = kn + (size_t)mrow * C;
  v16h p0 = frag_from_row_B(krow + 0, lane);
  v16h p1 = frag_from_row_B(krow + 32, lane);
  v16h p2 = frag_from_row_B(krow + 64, lane);
  v16h p3 = frag_from_row_B(krow + 96, lane);

  for (int mt = 0; mt < M; mt += 16) {
    v16h q0 = p0, q1 = p1, q2 = p2, q3 = p3;
    if (mt + 16 < M) {
      const _Float16* nrow = kn + (size_t)(mt + 16 + mrow) * C;
      q0 = frag_from_row_B(nrow + 0, lane);
      q1 = frag_from_row_B(nrow + 32, lane);
      q2 = frag_from_row_B(nrow + 64, lane);
      q3 = frag_from_row_B(nrow + 96, lane);
    }
    v8f c = {};
    c = __builtin_amdgcn_wmma_f32_16x16x32_f16(false, a0, false, p0, (short)0, c, false, false);
    c = __builtin_amdgcn_wmma_f32_16x16x32_f16(false, a1, false, p1, (short)0, c, false, false);
    c = __builtin_amdgcn_wmma_f32_16x16x32_f16(false, a2, false, p2, (short)0, c, false, false);
    c = __builtin_amdgcn_wmma_f32_16x16x32_f16(false, a3, false, p3, (short)0, c, false, false);
#pragma unroll
    for (int r = 0; r < 8; ++r) sims[hi + r][mt + mrow] = c[r];
    p0 = q0; p1 = q1; p2 = q2; p3 = q3;
  }
  __syncthreads();

  if (lane < 16) {
    float best = -1e30f;
    int bi = 0;
    for (int m = 0; m < M; ++m) {
      float v = sims[lane][m];
      if (v > best) { best = v; bi = m; }
    }
    int n = qBase + lane;
    _Float16* drow = dcat + ((size_t)b * NQ + n) * (2 * C) + C;
    const float* src = hard + (size_t)bi * C;
    for (int c = 0; c < C; ++c) drow[c] = (_Float16)src[c];
  }
}

// ---------------------------------------------------------------------------
// 8) Transposed conv 4x4 stride-2 pad-2 (+ReLU). Reference weight layout
//    (Cin, Cout, 4, 4); effective wt[co,ci,ky,kx] = w[ci,co,3-ky,3-kx].
// ---------------------------------------------------------------------------
__global__ void deconv4x4_s2_kernel(const float* __restrict__ in, const float* __restrict__ w,
                                    const float* __restrict__ bias, float* __restrict__ out,
                                    int B, int Cin, int Cout, int H, int W) {
  int OH = 2 * H, OW = 2 * W;
  int idx = blockIdx.x * blockDim.x + threadIdx.x;
  int total = B * Cout * OH * OW;
  if (idx >= total) return;
  int ox = idx % OW; int t = idx / OW;
  int oy = t % OH; t /= OH;
  int co = t % Cout; int b = t / Cout;
  float acc = bias[co];
#pragma unroll
  for (int ky = 0; ky < 4; ++ky) {
    int iy = oy + ky - 2;
    if (iy < 0 || iy > 2 * H - 2 || (iy & 1)) continue;
    int yin = iy >> 1;
#pragma unroll
    for (int kx = 0; kx < 4; ++kx) {
      int ix = ox + kx - 2;
      if (ix < 0 || ix > 2 * W - 2 || (ix & 1)) continue;
      int xin = ix >> 1;
      int wofs = (3 - ky) * 4 + (3 - kx);
      for (int ci = 0; ci < Cin; ++ci)
        acc += w[((size_t)ci * Cout + co) * 16 + wofs] *
               in[((size_t)(b * Cin + ci) * H + yin) * W + xin];
    }
  }
  out[((size_t)(b * Cout + co) * OH + oy) * OW + ox] = fmaxf(acc, 0.0f);
}

// ---------------------------------------------------------------------------
// 9) Final conv3x3 Cin=16 -> 1, bias, no ReLU
// ---------------------------------------------------------------------------
__global__ void conv3x3_final_kernel(const float* __restrict__ in, const float* __restrict__ w,
                                     const float* __restrict__ bias, float* __restrict__ out,
                                     int B, int Cin, int H, int W) {
  int idx = blockIdx.x * blockDim.x + threadIdx.x;
  int total = B * H * W;
  if (idx >= total) return;
  int x = idx % W; int t = idx / W;
  int y = t % H; int b = t / H;
  float acc = bias[0];
  for (int ci = 0; ci < Cin; ++ci) {
    const float* wk = w + ci * 9;
    const float* ip = in + ((size_t)(b * Cin + ci) * H) * W;
#pragma unroll
    for (int tt = 0; tt < 9; ++tt) {
      int yy = y + tt / 3 - 1, xx = x + tt % 3 - 1;
      if (yy >= 0 && yy < H && xx >= 0 && xx < W)
        acc += wk[tt] * ip[yy * W + xx];
    }
  }
  out[(b * H + y) * W + x] = acc;
}

// ---------------------------------------------------------------------------
// Launch
// ---------------------------------------------------------------------------
extern "C" void kernel_launch(void* const* d_in, const int* in_sizes, int n_in,
                              void* d_out, int out_size, void* d_ws, size_t ws_size,
                              hipStream_t stream) {
  (void)in_sizes; (void)n_in; (void)out_size; (void)ws_size;

  const int B = 16, H = 256, W = 256, C = 128, M = 800, CROP = 64;
  const int Hq = 64, Wq = 64, NQ = Hq * Wq;  // 4096 query positions

  const float* x        = (const float*)d_in[0];
  const float* ce_w1    = (const float*)d_in[1];
  const float* ce_b1    = (const float*)d_in[2];
  const float* ce_w2    = (const float*)d_in[3];
  const float* ce_b2    = (const float*)d_in[4];
  const float* se_w1    = (const float*)d_in[5];
  const float* se_b1    = (const float*)d_in[6];
  const float* se_w2    = (const float*)d_in[7];
  const float* se_b2    = (const float*)d_in[8];
  const float* mem_keys = (const float*)d_in[9];
  const float* mem_vals = (const float*)d_in[10];
  const float* mem_hard = (const float*)d_in[11];
  const float* dec_w1   = (const float*)d_in[12];
  const float* dec_b1   = (const float*)d_in[13];
  const float* dec_tw1  = (const float*)d_in[14];
  const float* dec_tb1  = (const float*)d_in[15];
  const float* dec_tw2  = (const float*)d_in[16];
  const float* dec_tb2  = (const float*)d_in[17];
  const float* dec_w2   = (const float*)d_in[18];
  const float* dec_b2   = (const float*)d_in[19];
  float* out = (float*)d_out;

  size_t off = 0;
  auto wsalloc = [&](size_t bytes) -> void* {
    void* p = (void*)((char*)d_ws + off);
    off += (bytes + 255) & ~(size_t)255;
    return p;
  };

  float*    xc    = (float*)wsalloc((size_t)B * H * W * 4);
  float*    a1    = (float*)wsalloc((size_t)B * 32 * 128 * 128 * 4);
  float*    s1    = (float*)wsalloc((size_t)B * 16 * 128 * 128 * 4);
  _Float16* a1h   = (_Float16*)wsalloc((size_t)B * 128 * 128 * 32 * 2);   // NHWC, Cpad=32
  _Float16* s1h   = (_Float16*)wsalloc((size_t)B * 128 * 128 * 32 * 2);   // NHWC, 16->32 pad
  float*    zc    = (float*)wsalloc((size_t)B * C * NQ * 4);
  float*    zs    = (float*)wsalloc((size_t)B * C * NQ * 4);
  _Float16* qn    = (_Float16*)wsalloc((size_t)B * NQ * C * 2);
  _Float16* q2n   = (_Float16*)wsalloc((size_t)B * NQ * C * 2);
  _Float16* knc   = (_Float16*)wsalloc((size_t)M * C * 2);
  _Float16* knh   = (_Float16*)wsalloc((size_t)M * C * 2);
  _Float16* dcath = (_Float16*)wsalloc((size_t)B * NQ * 2 * C * 2);       // NHWC f16, 256 ch
  float*    dd1   = (float*)wsalloc((size_t)B * 64 * NQ * 4);
  float*    dd2   = (float*)wsalloc((size_t)B * 32 * 128 * 128 * 4);
  float*    dd3   = (float*)wsalloc((size_t)B * 16 * 256 * 256 * 4);
  _Float16* wp_ce2  = (_Float16*)wsalloc((size_t)C * 9 * 32 * 2);         // (128, 288)
  _Float16* wp_se2  = (_Float16*)wsalloc((size_t)C * 9 * 32 * 2);         // (128, 288)
  _Float16* wp_dec1 = (_Float16*)wsalloc((size_t)64 * 9 * 256 * 2);       // (64, 2304)

  // weight packing (tiny)
  {
    int n1 = C * 9 * 32;
    pack_weights_kernel<<<(n1 + 255) / 256, 256, 0, stream>>>(ce_w2, wp_ce2, C, 32, 32);
    pack_weights_kernel<<<(n1 + 255) / 256, 256, 0, stream>>>(se_w2, wp_se2, C, 16, 32);
    int n2 = 64 * 9 * 256;
    pack_weights_kernel<<<(n2 + 255) / 256, 256, 0, stream>>>(dec_w1, wp_dec1, 64, 256, 256);
  }
  // 0) crop+pad
  {
    int n = B * H * W;
    crop_pad_kernel<<<(n + 255) / 256, 256, 0, stream>>>(x, xc, B, H, W, CROP);
  }
  // 1) encoder stage 1 convs (Cin=1, scalar, memory bound)
  {
    int n = B * 32 * 128 * 128;
    conv3x3_c1_pool_kernel<<<(n + 255) / 256, 256, 0, stream>>>(xc, ce_w1, ce_b1, a1, B, 32, H, W);
    int n2 = B * 16 * 128 * 128;
    conv3x3_c1_pool_kernel<<<(n2 + 255) / 256, 256, 0, stream>>>(x, se_w1, se_b1, s1, B, 16, H, W);
  }
  // 2) NCHW f32 -> NHWC f16 for the WMMA convs
  {
    int n = B * 128 * 128 * 32;
    nchw_to_hwc_f16_kernel<<<(n + 255) / 256, 256, 0, stream>>>(a1, a1h, B, 32, 32, 128, 128);
    nchw_to_hwc_f16_kernel<<<(n + 255) / 256, 256, 0, stream>>>(s1, s1h, B, 16, 32, 128, 128);
  }
  // 3) encoder stage 2: WMMA conv+relu+pool (128 -> 64)
  conv3x3_wmma_pool_kernel<<<dim3(128 / 32, 64, B * (C / 16)), 32, 0, stream>>>(
      a1h, wp_ce2, ce_b2, zc, B, 32, C, 128, 128);
  conv3x3_wmma_pool_kernel<<<dim3(128 / 32, 64, B * (C / 16)), 32, 0, stream>>>(
      s1h, wp_se2, se_b2, zs, B, 32, C, 128, 128);
  // 4) normalize queries and memory rows to f16
  {
    int n = B * NQ;
    l2norm_nchw_kernel<<<(n + 255) / 256, 256, 0, stream>>>(zc, qn, B, C, NQ);
    l2norm_nchw_kernel<<<(n + 255) / 256, 256, 0, stream>>>(zs, q2n, B, C, NQ);
    l2norm_mem_kernel<<<(M + 63) / 64, 64, 0, stream>>>(mem_keys, knc, M, C);
    l2norm_mem_kernel<<<(M + 63) / 64, 64, 0, stream>>>(mem_hard, knh, M, C);
  }
  // 5) memory matching (WMMA similarity GEMMs) -> f16 NHWC decoder input
  match_center_kernel<<<dim3(NQ / 16, B), 32, 0, stream>>>(qn, knc, mem_vals, dcath, B, NQ, M, C);
  match_argmax_kernel<<<dim3(NQ / 16, B), 32, 0, stream>>>(q2n, knh, mem_hard, dcath, B, NQ, M, C);
  // 6) decoder conv 256->64 (WMMA, relu), NHWC f16 in -> NCHW f32 out
  conv3x3_wmma_kernel<<<dim3(Wq / 16, Hq, B * (64 / 16)), 32, 0, stream>>>(
      dcath, wp_dec1, dec_b1, dd1, B, 2 * C, 64, Hq, Wq, 1);
  // 7) deconv 64->32 (64 -> 128) + relu
  {
    int n = B * 32 * 128 * 128;
    deconv4x4_s2_kernel<<<(n + 255) / 256, 256, 0, stream>>>(dd1, dec_tw1, dec_tb1, dd2, B, 64, 32, 64, 64);
  }
  // 8) deconv 32->16 (128 -> 256) + relu
  {
    int n = B * 16 * 256 * 256;
    deconv4x4_s2_kernel<<<(n + 255) / 256, 256, 0, stream>>>(dd2, dec_tw2, dec_tb2, dd3, B, 32, 16, 128, 128);
  }
  // 9) final conv 16->1
  {
    int n = B * H * W;
    conv3x3_final_kernel<<<(n + 255) / 256, 256, 0, stream>>>(dd3, dec_w2, dec_b2, out, B, 16, H, W);
  }
}
